// QuantizedConv1d_42580305772729
// MI455X (gfx1250) — 8-bit matrix operand order repaired and checked by reading, compiled, not run on hardware
//
#include <hip/hip_runtime.h>
#include <stdint.h>

typedef __attribute__((ext_vector_type(4))) int   v4i;
typedef __attribute__((ext_vector_type(8))) int   v8i;
typedef __attribute__((ext_vector_type(4))) float v4f;

#define B_    32
#define CIN_  128
#define L_    4096
#define COUT_ 256
#define KW_   5
#define PAD_  2
#define LP_   (L_ + 2*PAD_)

__global__ void __launch_bounds__(256)
quant_transpose_kernel(const float* __restrict__ x, int8_t* __restrict__ xqT) {
  __shared__ int8_t tile[64][68];
  const int b   = blockIdx.z;
  const int ci0 = blockIdx.y * 64;
  const int p0  = blockIdx.x * 64;
  const int tid = threadIdx.x;

  for (int i = tid; i < 64 * 64; i += 256) {
    int l  = i & 63;
    int ci = i >> 6;
    int src = p0 + l - PAD_;
    src = src < 0 ? 0 : (src > (L_ - 1) ? (L_ - 1) : src);
    float v = __builtin_nontemporal_load(
        x + ((size_t)(b * CIN_ + ci0 + ci)) * L_ + src);
    int q = (int)rintf(v * 20.0f) + (-3);
    q = q < -128 ? -128 : (q > 127 ? 127 : q);
    tile[l][ci] = (int8_t)q;
  }
  __syncthreads();

  for (int i = tid; i < 64 * 16; i += 256) {
    int c4 = (i & 15) * 4;
    int l  = i >> 4;
    int p  = p0 + l;
    if (p < LP_) {
      int v = *(const int*)&tile[l][c4];
      *(int*)&xqT[((size_t)(b * LP_ + p)) * CIN_ + ci0 + c4] = v;
    }
  }
}

__global__ void weight_pack_kernel(const int8_t* __restrict__ w,
                                   int8_t* __restrict__ wpack) {
  int idx = blockIdx.x * blockDim.x + threadIdx.x;
  if (idx >= KW_ * 2 * 16 * 32 * 8) return;
  int v    = idx & 7;
  int lane = (idx >> 3) & 31;
  int coT  = (idx >> 8) & 15;
  int c    = (idx >> 12) & 1;
  int k    = idx >> 13;

  int co   = coT * 16 + (lane & 15);
  int g    = v >> 1;
  int half = (lane >= 16) ? 16 : 0;
  uint32_t d = 0;
  for (int e = 0; e < 4; ++e) {
    int K  = (v >> 2) * 32 + half + (v & 3) * 4 + e;
    int ci = c * 64 + K;
    uint8_t byte = (uint8_t)w[((size_t)co * CIN_ + ci) * KW_ + k];
    d |= ((uint32_t)byte) << (8 * e);
  }
  ((uint32_t*)wpack)[idx] = d;
}

__global__ void __launch_bounds__(256)
conv_wmma_kernel(const int8_t* __restrict__ xqT, const int8_t* __restrict__ wpack,
                 const int* __restrict__ bias, const float* __restrict__ wscale,
                 float* __restrict__ out) {
  const int lane    = threadIdx.x & 31;
  const int waveId  = blockIdx.x * 8 + (threadIdx.x >> 5);
  const int coGroup = waveId & 3;
  const int lTile   = (waveId >> 2) & 255;
  const int b       = waveId >> 10;
  const int l0      = lTile * 16;
  const int coT0    = coGroup * 4;

  v8i acc[4];
  #pragma unroll
  for (int m = 0; m < 4; ++m) acc[m] = (v8i){0, 0, 0, 0, 0, 0, 0, 0};

  const int col         = lane & 15;
  const int ciStartHalf = (lane >= 16) ? 16 : 0;

  #pragma unroll
  for (int k = 0; k < KW_; ++k) {
    #pragma unroll
    for (int c = 0; c < 2; ++c) {
      const size_t rowBase =
          ((size_t)(b * LP_ + l0 + col + k)) * CIN_ + (size_t)(c * 64 + ciStartHalf);
      v4i blo = *(const v4i*)(xqT + rowBase);
      v4i bhi = *(const v4i*)(xqT + rowBase + 32);
      v8i bf;
      bf[0] = blo[0]; bf[1] = blo[1]; bf[2] = blo[2]; bf[3] = blo[3];
      bf[4] = bhi[0]; bf[5] = bhi[1]; bf[6] = bhi[2]; bf[7] = bhi[3];

      #pragma unroll
      for (int m = 0; m < 4; ++m) {
        const size_t aoff =
            ((size_t)(((k * 2 + c) * 16 + (coT0 + m)) * 32 + lane)) * 32;
        v4i alo = *(const v4i*)(wpack + aoff);
        v4i ahi = *(const v4i*)(wpack + aoff + 16);
        v8i af;
        af[0] = alo[0]; af[1] = alo[1]; af[2] = alo[2]; af[3] = alo[3];
        af[4] = ahi[0]; af[5] = ahi[1]; af[6] = ahi[2]; af[7] = ahi[3];
        acc[m] = __builtin_amdgcn_wmma_i32_16x16x64_iu8(true, af, true, bf,
                                                        acc[m], false, false);
      }
    }
  }

  const int n        = lane & 15;
  const int mrowHalf = (lane >= 16) ? 8 : 0;
  #pragma unroll
  for (int m = 0; m < 4; ++m) {
    const int cobase = (coT0 + m) * 16 + mrowHalf;
    v4i bia0 = *(const v4i*)(bias + cobase);
    v4i bia1 = *(const v4i*)(bias + cobase + 4);
    v4f sc0  = *(const v4f*)(wscale + cobase);
    v4f sc1  = *(const v4f*)(wscale + cobase + 4);
    #pragma unroll
    for (int v = 0; v < 8; ++v) {
      int   bb = (v < 4) ? bia0[v] : bia1[v - 4];
      float ss = (v < 4) ? sc0[v]  : sc1[v - 4];
      int   a  = acc[m][v] + bb;
      float r  = rintf((float)a * (0.5f * ss) + (-128.0f));
      r = r < -128.0f ? -128.0f : (r > 127.0f ? 127.0f : r);
      __builtin_nontemporal_store(
          r, out + ((size_t)(b * COUT_ + cobase + v)) * L_ + l0 + n);
    }
  }
}

extern "C" void kernel_launch(void* const* d_in, const int* in_sizes, int n_in,
                              void* d_out, int out_size, void* d_ws, size_t ws_size,
                              hipStream_t stream) {
  (void)in_sizes; (void)n_in; (void)out_size; (void)ws_size;
  const float*  x      = (const float*)d_in[0];
  const int8_t* w      = (const int8_t*)d_in[1];
  const int*    bias   = (const int*)d_in[2];
  const float*  wscale = (const float*)d_in[3];
  float*        out    = (float*)d_out;

  int8_t* xqT   = (int8_t*)d_ws;
  int8_t* wpack = xqT + (size_t)B_ * LP_ * CIN_;

  dim3 qgrid((LP_ + 63) / 64, CIN_ / 64, B_);
  quant_transpose_kernel<<<qgrid, 256, 0, stream>>>(x, xqT);

  const int packN = KW_ * 2 * 16 * 32 * 8;
  weight_pack_kernel<<<(packN + 255) / 256, 256, 0, stream>>>(w, wpack);

  conv_wmma_kernel<<<4096, 256, 0, stream>>>(xqT, wpack, bias, wscale, out);
}
